// RealNVP_73263552135281
// MI455X (gfx1250) — matrix operand order repaired and checked by reading, not compiled, not run on hardware
//
#include <hip/hip_runtime.h>
#include <stdint.h>

#ifndef USE_ASYNC_LDS
#define USE_ASYNC_LDS 1
#endif

typedef __attribute__((ext_vector_type(16))) _Float16 v16h;
typedef __attribute__((ext_vector_type(8)))  _Float16 v8h;
typedef __attribute__((ext_vector_type(4)))  _Float16 v4h;
typedef __attribute__((ext_vector_type(8)))  float    v8f;

#define LAYERS 6
#define HID    128
#define BATCH  262144
#define ALPHA  0.6f

// workspace layout (bytes)
#define OFF_W1   0        // f16 [L][8 tiles][16 cols][4]          = 6144 B
#define OFF_W2   8192     // f16 [L][4 chunks][8 tiles][32 ln][16] = 196608 B
#define OFF_W3   204800   // f16 [L][4 chunks][32 ln][16]          = 24576 B
#define OFF_ANS  229376   // f32 [L][8] exp(clipped log_scale)
#define OFF_SUM  229600   // f32 [L]   sum(clipped log_scale)
#define OFF_ELG  229664   // f32 [L][4] exp(clipped log_gamma)

union V16u { v16h v; v8h h[2]; };

__device__ __forceinline__ v8f wmma_f16(v16h a, v16h b, v8f c) {
  return __builtin_amdgcn_wmma_f32_16x16x32_f16(false, a, false, b, (short)0, c, false, false);
}

__device__ __forceinline__ float fast_tanh(float x) {
  float ax = fabsf(x);
  float t  = __expf(-2.0f * ax);
  float r  = (1.0f - t) / (1.0f + t);
  return copysignf(r, x);
}

// ---------------- weight prep: f32 -> f16 in WMMA B-fragment order ----------------
__global__ void prep_kernel(const float* __restrict__ anls,
                            const float* __restrict__ W1,
                            const float* __restrict__ W2,
                            const float* __restrict__ W3,
                            const float* __restrict__ lg,
                            char* __restrict__ ws)
{
  _Float16* w1o = (_Float16*)(ws + OFF_W1);
  _Float16* w2o = (_Float16*)(ws + OFF_W2);
  _Float16* w3o = (_Float16*)(ws + OFF_W3);
  float* anso = (float*)(ws + OFF_ANS);
  float* sumo = (float*)(ws + OFF_SUM);
  float* elgo = (float*)(ws + OFF_ELG);
  const int tid    = blockIdx.x * blockDim.x + threadIdx.x;
  const int stride = gridDim.x * blockDim.x;

  // W2 fragments: element e of lane ln in frag (ly,c,t) = W2[ly][32c+16*half+e][16t+lc]
  for (int idx = tid; idx < LAYERS * 4 * 8 * 32 * 16; idx += stride) {
    int e = idx & 15, ln = (idx >> 4) & 31, t = (idx >> 9) & 7, c = (idx >> 12) & 3, ly = idx >> 14;
    int hh = ln >> 4, lcc = ln & 15;
    int k = 32 * c + 16 * hh + e, n = 16 * t + lcc;
    w2o[idx] = (_Float16)W2[(ly * HID + k) * HID + n];
  }
  // W1 compact (only K=0..3 nonzero): [(ly,t,col)][j] = W1[ly][j][16t+col]
  for (int idx = tid; idx < LAYERS * 8 * 16 * 4; idx += stride) {
    int j = idx & 3, colv = (idx >> 2) & 15, t = (idx >> 6) & 7, ly = idx >> 9;
    w1o[idx] = (_Float16)W1[(ly * 4 + j) * HID + 16 * t + colv];
  }
  // W3 padded to 128x16 (cols 8..15 zero), fragment order
  for (int idx = tid; idx < LAYERS * 4 * 32 * 16; idx += stride) {
    int e = idx & 15, ln = (idx >> 4) & 31, c = (idx >> 9) & 3, ly = idx >> 11;
    int hh = ln >> 4, lcc = ln & 15;
    int k = 32 * c + 16 * hh + e;
    w3o[idx] = (lcc < 8) ? (_Float16)W3[(ly * HID + k) * 8 + lcc] : (_Float16)0.0f;
  }
  for (int idx = tid; idx < LAYERS * 8; idx += stride) {
    float cl = fminf(fmaxf(anls[idx], -5.0f), 5.0f);
    anso[idx] = expf(cl);
  }
  for (int idx = tid; idx < LAYERS; idx += stride) {
    float s = 0.0f;
    for (int d = 0; d < 8; d++) s += fminf(fmaxf(anls[idx * 8 + d], -5.0f), 5.0f);
    sumo[idx] = s;
  }
  for (int idx = tid; idx < LAYERS * 4; idx += stride) {
    elgo[idx] = expf(fminf(fmaxf(lg[idx], -5.0f), 5.0f));
  }
}

// ---------------- fused flow kernel ----------------
__global__ __launch_bounds__(256) void flow_kernel(
    const float* __restrict__ x,
    const float* __restrict__ anb,
    const float* __restrict__ b1g,
    const float* __restrict__ b2g,
    const float* __restrict__ b3g,
    const char*  __restrict__ ws,
    float* __restrict__ out)
{
  __shared__ _Float16 hbuf[8][32][HID];        // 64 KB: per-wave h staging / tail scratch
  __shared__ _Float16 w2s[4][8][32][16];       // 32 KB: one layer of W2 fragments
  __shared__ float pB1[LAYERS][HID];
  __shared__ float pB2[LAYERS][HID];
  __shared__ float pB3[LAYERS][8];
  __shared__ float pAnb[LAYERS][8];
  __shared__ float pAns[LAYERS][8];
  __shared__ float pSum[LAYERS];
  __shared__ float pElg[LAYERS][4];

  const _Float16* w1g = (const _Float16*)(ws + OFF_W1);
  const char*     w2g = ws + OFF_W2;
  const _Float16* w3g = (const _Float16*)(ws + OFF_W3);
  const float* ansg = (const float*)(ws + OFF_ANS);
  const float* sumg = (const float*)(ws + OFF_SUM);
  const float* elgg = (const float*)(ws + OFF_ELG);

  const int tid  = threadIdx.x;
  const int wave = tid >> 5;
  const int lane = tid & 31;
  const int h16  = lane >> 4;
  const int lc   = lane & 15;
  const _Float16 zh = (_Float16)0.0f;

  for (int k = tid; k < LAYERS * HID; k += 256) {
    (&pB1[0][0])[k] = b1g[k];
    (&pB2[0][0])[k] = b2g[k];
  }
  for (int k = tid; k < LAYERS * 8; k += 256) {
    (&pB3[0][0])[k]  = b3g[k];
    (&pAnb[0][0])[k] = anb[k];
    (&pAns[0][0])[k] = ansg[k];
  }
  if (tid < LAYERS)     pSum[tid] = sumg[tid];
  if (tid < LAYERS * 4) (&pElg[0][0])[tid] = elgg[tid];

  // prefetch W2 layer 0 into LDS (async tensor-ish path: ASYNCcnt)
  {
    const char* src = w2g + tid * 16;
    unsigned dst = (unsigned)(uint64_t)((const char*)&w2s[0][0][0][0] + tid * 16);
#pragma unroll
    for (int j = 0; j < 8; j++) {
#if USE_ASYNC_LDS
      asm volatile("global_load_async_to_lds_b128 %0, %1, off"
                   :: "v"(dst + j * 4096), "v"(src + j * 4096) : "memory");
#else
      *(float4*)((char*)&w2s[0][0][0][0] + tid * 16 + j * 4096) = *(const float4*)(src + j * 4096);
#endif
    }
  }
  __syncthreads(); // params visible

  const long row = (long)blockIdx.x * 256 + tid;   // lane owns one row
  float z[8];
  {
    const float4* xp = (const float4*)(x + row * 8);
    float4 a = xp[0], b = xp[1];
    z[0] = a.x; z[1] = a.y; z[2] = a.z; z[3] = a.w;
    z[4] = b.x; z[5] = b.y; z[6] = b.z; z[7] = b.w;
  }
  float logdet = 0.0f;

  _Float16* hb = &hbuf[wave][0][0];

  for (int i = 0; i < LAYERS; i++) {
    const bool odd = (i & 1) != 0;

    // ---- activation norm ----
#pragma unroll
    for (int d = 0; d < 8; d++) z[d] = (z[d] + pAnb[i][d]) * pAns[i][d];
    logdet += pSum[i];

    // ---- GEMM1: h1 = relu(xf @ W1 + b1), K=4 padded to 32 ----
    float f0 = odd ? z[1] : z[0];
    float f1 = odd ? z[3] : z[2];
    float f2 = odd ? z[5] : z[4];
    float f3 = odd ? z[7] : z[6];
    float g0 = __shfl(f0, lc + 16, 32);  // rows 16..31 for second M-tile
    float g1 = __shfl(f1, lc + 16, 32);
    float g2 = __shfl(f2, lc + 16, 32);
    float g3 = __shfl(f3, lc + 16, 32);
    v16h a0, a1;
#pragma unroll
    for (int e = 0; e < 16; e++) { a0[e] = zh; a1[e] = zh; }
    a0[0] = h16 ? zh : (_Float16)f0; a0[1] = h16 ? zh : (_Float16)f1;
    a0[2] = h16 ? zh : (_Float16)f2; a0[3] = h16 ? zh : (_Float16)f3;
    a1[0] = h16 ? zh : (_Float16)g0; a1[1] = h16 ? zh : (_Float16)g1;
    a1[2] = h16 ? zh : (_Float16)g2; a1[3] = h16 ? zh : (_Float16)g3;

#pragma unroll
    for (int t = 0; t < 8; t++) {
      const int col = 16 * t + lc;
      float bias = pB1[i][col];
      v8f c0, c1;
#pragma unroll
      for (int r = 0; r < 8; r++) { c0[r] = bias; c1[r] = bias; }
      v4h w = *(const v4h*)(w1g + (((i * 8 + t) * 16) + lc) * 4);
      v16h bfr;
#pragma unroll
      for (int e = 0; e < 16; e++) bfr[e] = zh;
      bfr[0] = h16 ? zh : w[0]; bfr[1] = h16 ? zh : w[1];
      bfr[2] = h16 ? zh : w[2]; bfr[3] = h16 ? zh : w[3];
      c0 = wmma_f16(a0, bfr, c0);
      c1 = wmma_f16(a1, bfr, c1);
#pragma unroll
      for (int r = 0; r < 8; r++) {
        hb[(r + 8 * h16) * HID + col]      = (_Float16)fmaxf(c0[r], 0.0f);
        hb[(16 + r + 8 * h16) * HID + col] = (_Float16)fmaxf(c1[r], 0.0f);
      }
    }
    asm volatile("" ::: "memory");

    // W2 of this layer must be in LDS now
#if USE_ASYNC_LDS
    asm volatile("s_wait_asynccnt 0" ::: "memory");
#endif
    __syncthreads();

    // load h1 as A fragments (kept resident; hbuf becomes reusable)
    v16h A2[2][4];
#pragma unroll
    for (int s = 0; s < 2; s++) {
#pragma unroll
      for (int c = 0; c < 4; c++) {
        const int m = 16 * s + lc;
        const int kb = 32 * c + 16 * h16;
        V16u u;
        u.h[0] = *(const v8h*)&hb[m * HID + kb];
        u.h[1] = *(const v8h*)&hb[m * HID + kb + 8];
        A2[s][c] = u.v;
      }
    }
    asm volatile("" ::: "memory");

    // ---- GEMM2: h2 = relu(h1 @ W2 + b2) ----
#pragma unroll
    for (int t = 0; t < 8; t++) {
      const int col = 16 * t + lc;
      float bias = pB2[i][col];
      v8f c0, c1;
#pragma unroll
      for (int r = 0; r < 8; r++) { c0[r] = bias; c1[r] = bias; }
#pragma unroll
      for (int c = 0; c < 4; c++) {
        const _Float16* bp = &w2s[c][t][lane][0];
        V16u u;
        u.h[0] = *(const v8h*)bp;
        u.h[1] = *(const v8h*)(bp + 8);
        c0 = wmma_f16(A2[0][c], u.v, c0);
        c1 = wmma_f16(A2[1][c], u.v, c1);
      }
#pragma unroll
      for (int r = 0; r < 8; r++) {
        hb[(r + 8 * h16) * HID + col]      = (_Float16)fmaxf(c0[r], 0.0f);
        hb[(16 + r + 8 * h16) * HID + col] = (_Float16)fmaxf(c1[r], 0.0f);
      }
    }
    asm volatile("" ::: "memory");
    __syncthreads();  // all waves finished reading w2s

    // prefetch next layer W2 (overlaps GEMM3/tail/next GEMM1)
    if (i + 1 < LAYERS) {
      const char* src = w2g + (i + 1) * 32768 + tid * 16;
      unsigned dst = (unsigned)(uint64_t)((const char*)&w2s[0][0][0][0] + tid * 16);
#pragma unroll
      for (int j = 0; j < 8; j++) {
#if USE_ASYNC_LDS
        asm volatile("global_load_async_to_lds_b128 %0, %1, off"
                     :: "v"(dst + j * 4096), "v"(src + j * 4096) : "memory");
#else
        *(float4*)((char*)&w2s[0][0][0][0] + tid * 16 + j * 4096) = *(const float4*)(src + j * 4096);
#endif
      }
    }

    // ---- GEMM3: h3 = h2 @ W3 + b3 (N padded 8->16) ----
    v16h A3[2][4];
#pragma unroll
    for (int s = 0; s < 2; s++) {
#pragma unroll
      for (int c = 0; c < 4; c++) {
        const int m = 16 * s + lc;
        const int kb = 32 * c + 16 * h16;
        V16u u;
        u.h[0] = *(const v8h*)&hb[m * HID + kb];
        u.h[1] = *(const v8h*)&hb[m * HID + kb + 8];
        A3[s][c] = u.v;
      }
    }
    asm volatile("" ::: "memory");
    float bias3 = (lc < 8) ? pB3[i][lc] : 0.0f;
    v8f d0, d1;
#pragma unroll
    for (int r = 0; r < 8; r++) { d0[r] = bias3; d1[r] = bias3; }
#pragma unroll
    for (int c = 0; c < 4; c++) {
      const _Float16* bp = w3g + (((i * 4 + c) * 32) + lane) * 16;
      V16u u;
      u.h[0] = *(const v8h*)bp;
      u.h[1] = *(const v8h*)(bp + 8);
      d0 = wmma_f16(A3[0][c], u.v, d0);
      d1 = wmma_f16(A3[1][c], u.v, d1);
    }

    // ---- tail: redistribute D so each lane holds its own row's 8 outputs ----
    float* tb = (float*)&hbuf[wave][0][0];
    if (lc < 8) {
#pragma unroll
      for (int r = 0; r < 8; r++) {
        tb[(r + 8 * h16) * 8 + lc]      = d0[r];
        tb[(16 + r + 8 * h16) * 8 + lc] = d1[r];
      }
    }
    asm volatile("" ::: "memory");
    float4 s0 = *(const float4*)&tb[lane * 8];
    float4 s1 = *(const float4*)&tb[lane * 8 + 4];
    float sraw[8] = {s0.x, s0.y, s0.z, s0.w, s1.x, s1.y, s1.z, s1.w};

    float ush[4], usc[4];
#pragma unroll
    for (int tt = 0; tt < 4; tt++) {
      ush[tt] = pElg[i][tt] * fast_tanh(sraw[tt]);
      float sc = ALPHA * fast_tanh(sraw[4 + tt]);
      usc[tt] = 1.0f + sc;
      logdet += __logf(fabsf(usc[tt]) + 1e-8f);
    }
    if (odd) {  // ti = {0,2,4,6}
      z[0] = z[0] * usc[0] + ush[0]; z[2] = z[2] * usc[1] + ush[1];
      z[4] = z[4] * usc[2] + ush[2]; z[6] = z[6] * usc[3] + ush[3];
    } else {    // ti = {1,3,5,7}
      z[1] = z[1] * usc[0] + ush[0]; z[3] = z[3] * usc[1] + ush[1];
      z[5] = z[5] * usc[2] + ush[2]; z[7] = z[7] * usc[3] + ush[3];
    }
    // physical reversal z = z[::-1]
#pragma unroll
    for (int d = 0; d < 4; d++) { float tmp = z[d]; z[d] = z[7 - d]; z[7 - d] = tmp; }
    asm volatile("" ::: "memory");
  }

  float4 o0, o1;
  o0.x = z[0]; o0.y = z[1]; o0.z = z[2]; o0.w = z[3];
  o1.x = z[4]; o1.y = z[5]; o1.z = z[6]; o1.w = z[7];
  float4* op = (float4*)(out + row * 8);
  op[0] = o0; op[1] = o1;
  out[(long)BATCH * 8 + row] = logdet;
}

extern "C" void kernel_launch(void* const* d_in, const int* in_sizes, int n_in,
                              void* d_out, int out_size, void* d_ws, size_t ws_size,
                              hipStream_t stream) {
  const float* x    = (const float*)d_in[0];
  const float* anls = (const float*)d_in[1];
  const float* anbi = (const float*)d_in[2];
  const float* W1   = (const float*)d_in[3];
  const float* b1   = (const float*)d_in[4];
  const float* W2   = (const float*)d_in[5];
  const float* b2   = (const float*)d_in[6];
  const float* W3   = (const float*)d_in[7];
  const float* b3   = (const float*)d_in[8];
  const float* lg   = (const float*)d_in[9];
  char*  ws  = (char*)d_ws;
  float* out = (float*)d_out;

  prep_kernel<<<dim3(96), dim3(256), 0, stream>>>(anls, W1, W2, W3, lg, ws);
  flow_kernel<<<dim3(BATCH / 256), dim3(256), 0, stream>>>(x, anbi, b1, b2, b3, ws, out);
}
